// DANetAttention_81681688036028
// MI455X (gfx1250) — matrix operand order repaired and checked by reading, not compiled, not run on hardware
//
#include <hip/hip_runtime.h>

// ---------------- constants ----------------
#define CC   512
#define NN   4096
#define DD   64      // C/8
#define BB   4

typedef __attribute__((ext_vector_type(16))) __bf16 bf16x16;
typedef __attribute__((ext_vector_type(8)))  float  f32x8;

union FragU { uint4 u[2]; bf16x16 v; };

// Compiler-only memory barrier: per-wave LDS ops are executed in order by HW
// (DScnt, in-order pipe); this just stops clang from reordering them.
#define WAVE_MEM_BARRIER() asm volatile("" ::: "memory")

// A-fragment (16x32 bf16, row-major source, K contiguous):
// lane L: row = m0 + (L&15); ko = (L<16)?0:8
// elems 0..7  = K [k+ko   .. k+ko+7 ]
// elems 8..15 = K [k+16+ko.. k+23+ko]
__device__ __forceinline__ bf16x16 load_frag_a(const __bf16* A, int lda, int m0, int k, int lane) {
  int row = m0 + (lane & 15);
  int ko  = (lane < 16) ? 0 : 8;
  const __bf16* p = A + (size_t)row * lda + k + ko;
  FragU f;
  f.u[0] = *(const uint4*)(p);
  f.u[1] = *(const uint4*)(p + 16);
  return f.v;
}

// B-fragment (32x16 bf16) sourced from Bt[n][k] row-major (K contiguous):
// lane L: col n (passed in); klo = (L<16)?0:8; elems 0..7 = K k+klo.., elems 8..15 = K k+16+klo..
__device__ __forceinline__ bf16x16 load_frag_b(const __bf16* Bt, int ldb, int n, int k, int lane) {
  int klo = (lane < 16) ? 0 : 8;
  const __bf16* p = Bt + (size_t)n * ldb + k + klo;
  FragU f;
  f.u[0] = *(const uint4*)(p);
  f.u[1] = *(const uint4*)(p + 16);
  return f.v;
}

__device__ __forceinline__ f32x8 wmma_bf16(bf16x16 a, bf16x16 b, f32x8 c) {
  return __builtin_amdgcn_wmma_f32_16x16x32_bf16(false, a, false, b, (short)0, c, false, false);
}

// ---------------- kernel 1: weight concat + bf16 convert ----------------
__global__ __launch_bounds__(256) void prep_weights(
    const float* Wq, const float* Wk, const float* Wsv,
    const float* Wv, const float* Wsq, const float* Wsk,
    const float* bq, const float* bk, const float* bsv,
    const float* bv, const float* bsq, const float* bsk,
    __bf16* Wcat1, __bf16* Wcat2, float* bias1, float* bias2) {
  int gid = blockIdx.x * 256 + threadIdx.x;
  const int W1 = 1536 * 512, W2 = 640 * 512;
  if (gid < W1) {
    int r = gid >> 9;
    float v = (r < 512) ? Wq[gid] : (r < 1024) ? Wk[gid - 512 * 512] : Wsv[gid - 1024 * 512];
    Wcat1[gid] = (__bf16)v;
  } else if (gid < W1 + W2) {
    int g = gid - W1;
    int r = g >> 9;
    float v = (r < 512) ? Wv[g] : (r < 576) ? Wsq[g - 512 * 512] : Wsk[g - 576 * 512];
    Wcat2[g] = (__bf16)v;
  }
  if (gid < 1536) {
    bias1[gid] = (gid < 512) ? bq[gid] : (gid < 1024) ? bk[gid - 512] : bsv[gid - 1024];
  } else if (gid < 1536 + 640) {
    int g = gid - 1536;
    bias2[g] = (g < 512) ? bv[g] : (g < 576) ? bsq[g - 512] : bsk[g - 576];
  }
}

// ---------------- kernel 2: x [B][C][N] f32 -> xT [B][N][C] bf16 ----------------
__global__ __launch_bounds__(256) void transpose_to_bf16(const float* __restrict__ x,
                                                         __bf16* __restrict__ xt) {
  __shared__ float tile[32][33];
  int b  = blockIdx.z;
  int n0 = blockIdx.x * 32;
  int c0 = blockIdx.y * 32;
  const float* xb = x + (size_t)b * CC * NN;
  #pragma unroll
  for (int i = 0; i < 4; ++i)
    tile[threadIdx.y + 8 * i][threadIdx.x] = xb[(size_t)(c0 + threadIdx.y + 8 * i) * NN + n0 + threadIdx.x];
  __syncthreads();
  __bf16* ob = xt + (size_t)b * NN * CC;
  #pragma unroll
  for (int i = 0; i < 4; ++i)
    ob[(size_t)(n0 + threadIdx.y + 8 * i) * CC + c0 + threadIdx.x] = (__bf16)tile[threadIdx.x][threadIdx.y + 8 * i];
}

// ---------------- generic NT GEMM: Y[M][Nn] = A[M][K] * Bt[Nn][K]^T (+bias) ----------------
// Wave tile: 32 (rows) x 64 (cols), 8 accumulators, B-fragments shared by both A rows.
// MODE 0: bf16 out, 1: f32 out, 2: f32 out = gamma*acc + 2*x (fused residual epilogue)
template <int MODE>
__global__ __launch_bounds__(256) void gemm_nt(
    const __bf16* __restrict__ A, const __bf16* __restrict__ Bt, void* __restrict__ Cout,
    int M, int Nn, int K, int lda, int ldb, int ldc,
    long long strideA, long long strideB, long long strideC,
    const float* __restrict__ bias, int bias_per_row,
    const float* __restrict__ xres, const float* __restrict__ gamma) {
  int lane = threadIdx.x & 31;
  int wave = threadIdx.x >> 5;
  int mt_count = M >> 5;
  int gw = blockIdx.x * 8 + wave;
  int mt = gw % mt_count;
  int nt = gw / mt_count;
  if (nt * 64 >= Nn) return;
  int b = blockIdx.y;
  const __bf16* Ab = A + (size_t)b * strideA;
  const __bf16* Bb = Bt + (size_t)b * strideB;
  int m0 = mt * 32, n0 = nt * 64;

  f32x8 acc[2][4] = {};
  for (int k = 0; k < K; k += 32) {
    bf16x16 a0 = load_frag_a(Ab, lda, m0, k, lane);
    bf16x16 a1 = load_frag_a(Ab, lda, m0 + 16, k, lane);
    #pragma unroll
    for (int cc = 0; cc < 4; ++cc) {
      bf16x16 bf = load_frag_b(Bb, ldb, n0 + cc * 16 + (lane & 15), k, lane);
      acc[0][cc] = wmma_bf16(a0, bf, acc[0][cc]);
      acc[1][cc] = wmma_bf16(a1, bf, acc[1][cc]);
    }
  }

  int colb  = lane & 15;
  int rbase = (lane < 16) ? 0 : 8;
  float g = (MODE == 2) ? gamma[0] : 0.f;
  #pragma unroll
  for (int h = 0; h < 2; ++h) {
    #pragma unroll
    for (int cc = 0; cc < 4; ++cc) {
      int col = n0 + cc * 16 + colb;
      #pragma unroll
      for (int r = 0; r < 8; ++r) {
        int row = m0 + h * 16 + rbase + r;
        float v = acc[h][cc][r];
        if (bias) v += bias_per_row ? bias[row] : bias[col];
        size_t idx = (size_t)row * ldc + col;
        if (MODE == 0) {
          ((__bf16*)Cout)[(size_t)b * strideC + idx] = (__bf16)v;
        } else if (MODE == 1) {
          ((float*)Cout)[(size_t)b * strideC + idx] = v;
        } else {
          float xv = xres[(size_t)b * strideC + idx];
          ((float*)Cout)[(size_t)b * strideC + idx] = g * v + 2.0f * xv;
        }
      }
    }
  }
}

// ---------------- kernel: row softmax on E [B*512][512] f32 -> bf16 ----------------
__global__ __launch_bounds__(256) void softmax_rows(const float* __restrict__ E,
                                                    __bf16* __restrict__ attn) {
  int row  = blockIdx.x * 8 + (threadIdx.x >> 5);
  int lane = threadIdx.x & 31;
  const float* e = E + (size_t)row * 512;
  float vals[16];
  float m = -3.0e38f;
  #pragma unroll
  for (int i = 0; i < 16; ++i) { vals[i] = e[lane + i * 32]; m = fmaxf(m, vals[i]); }
  #pragma unroll
  for (int off = 16; off >= 1; off >>= 1) m = fmaxf(m, __shfl_xor(m, off, 32));
  float s = 0.f;
  #pragma unroll
  for (int i = 0; i < 16; ++i) { vals[i] = __expf(vals[i] - m); s += vals[i]; }
  #pragma unroll
  for (int off = 16; off >= 1; off >>= 1) s += __shfl_xor(s, off, 32);
  float inv = 1.0f / s;
  __bf16* o = attn + (size_t)row * 512;
  #pragma unroll
  for (int i = 0; i < 16; ++i) o[lane + i * 32] = (__bf16)(vals[i] * inv);
}

// ---------------- flash spatial attention (two-pass, no per-iter shuffles) ----------------
// Q = sqT [N][64], K = skT [N][64], V[m][c] = sv[c][m] (sv is [C][N], K contiguous for B-frags)
// Each block: 16 query rows; wave w handles V columns [w*64, w*64+64)
// d_out[b][c][n] += gamma_s * O[n][c]  (single writer per element)
__device__ __forceinline__ void score_tiles(const __bf16* skT, int m0, int lane,
                                            bf16x16 qa0, bf16x16 qa1,
                                            f32x8& s0, f32x8& s1) {
  bf16x16 kb;
  s0 = (f32x8){};
  s1 = (f32x8){};
  kb = load_frag_b(skT, 640, m0 + (lane & 15), 0, lane);       s0 = wmma_bf16(qa0, kb, s0);
  kb = load_frag_b(skT, 640, m0 + (lane & 15), 32, lane);      s0 = wmma_bf16(qa1, kb, s0);
  kb = load_frag_b(skT, 640, m0 + 16 + (lane & 15), 0, lane);  s1 = wmma_bf16(qa0, kb, s1);
  kb = load_frag_b(skT, 640, m0 + 16 + (lane & 15), 32, lane); s1 = wmma_bf16(qa1, kb, s1);
}

__global__ __launch_bounds__(256) void flash_spatial(const __bf16* __restrict__ Y2,
                                                     const __bf16* __restrict__ Y1,
                                                     const float* __restrict__ gamma_s,
                                                     float* __restrict__ out) {
  int b    = blockIdx.y;
  int lane = threadIdx.x & 31;
  int wave = threadIdx.x >> 5;
  int n0   = blockIdx.x * 16;
  int c0   = wave * 64;

  const __bf16* sqT = Y2 + (size_t)b * NN * 640 + 512;   // [N][64], ld 640
  const __bf16* skT = Y2 + (size_t)b * NN * 640 + 576;   // [N][64], ld 640
  const __bf16* sv  = Y1 + (size_t)b * 1536 * NN + (size_t)1024 * NN;  // [C][N]

  __shared__ __align__(16) __bf16 pld[8][16 * 32];

  bf16x16 qa0 = load_frag_a(sqT, 640, n0, 0, lane);
  bf16x16 qa1 = load_frag_a(sqT, 640, n0, 32, lane);

  // ---- pass 1: row max (lane-local accumulation, one cross-lane reduce at end) ----
  float mloc[8];
  #pragma unroll
  for (int r = 0; r < 8; ++r) mloc[r] = -3.0e38f;
  for (int m0 = 0; m0 < NN; m0 += 32) {
    f32x8 s0, s1;
    score_tiles(skT, m0, lane, qa0, qa1, s0, s1);
    #pragma unroll
    for (int r = 0; r < 8; ++r) mloc[r] = fmaxf(mloc[r], fmaxf(s0[r], s1[r]));
  }
  #pragma unroll
  for (int r = 0; r < 8; ++r) {
    #pragma unroll
    for (int off = 8; off >= 1; off >>= 1) mloc[r] = fmaxf(mloc[r], __shfl_xor(mloc[r], off, 32));
  }

  // ---- pass 2: P = exp(S - mrow); O += P*V; lane-local denominator ----
  f32x8 o[4] = {};
  float ssum[8];
  #pragma unroll
  for (int r = 0; r < 8; ++r) ssum[r] = 0.f;
  int rowoff = (lane < 16) ? 0 : 8;
  int col    = lane & 15;

  for (int m0 = 0; m0 < NN; m0 += 32) {
    f32x8 s0, s1;
    score_tiles(skT, m0, lane, qa0, qa1, s0, s1);
    #pragma unroll
    for (int r = 0; r < 8; ++r) {
      float p0 = __expf(s0[r] - mloc[r]);
      float p1 = __expf(s1[r] - mloc[r]);
      ssum[r] += p0 + p1;
      int row = r + rowoff;
      pld[wave][row * 32 + col]      = (__bf16)p0;
      pld[wave][row * 32 + col + 16] = (__bf16)p1;
    }
    WAVE_MEM_BARRIER();  // per-wave LDS pipe is in-order; stop compiler reordering only
    bf16x16 pa;
    {
      int prow = lane & 15;
      int ko   = (lane < 16) ? 0 : 8;
      const __bf16* base = &pld[wave][prow * 32 + ko];
      FragU f;
      f.u[0] = *(const uint4*)(base);
      f.u[1] = *(const uint4*)(base + 16);
      pa = f.v;
    }
    WAVE_MEM_BARRIER();
    #pragma unroll
    for (int cc = 0; cc < 4; ++cc) {
      bf16x16 vb = load_frag_b(sv, NN, c0 + cc * 16 + (lane & 15), m0, lane);
      o[cc] = wmma_bf16(pa, vb, o[cc]);
    }
  }

  // one cross-lane reduce of the denominator
  #pragma unroll
  for (int r = 0; r < 8; ++r) {
    #pragma unroll
    for (int off = 8; off >= 1; off >>= 1) ssum[r] += __shfl_xor(ssum[r], off, 32);
  }

  float g = gamma_s[0];
  float* ob = out + (size_t)b * CC * NN;
  #pragma unroll
  for (int r = 0; r < 8; ++r) {
    float inv = g / ssum[r];
    int n = n0 + r + rowoff;
    #pragma unroll
    for (int cc = 0; cc < 4; ++cc) {
      int c = c0 + cc * 16 + (lane & 15);
      size_t idx = (size_t)c * NN + n;
      ob[idx] += o[cc][r] * inv;
    }
  }
}

// ---------------- host launch ----------------
extern "C" void kernel_launch(void* const* d_in, const int* in_sizes, int n_in,
                              void* d_out, int out_size, void* d_ws, size_t ws_size,
                              hipStream_t stream) {
  const float* x      = (const float*)d_in[0];
  const float* Wq     = (const float*)d_in[1];
  const float* bq     = (const float*)d_in[2];
  const float* Wk     = (const float*)d_in[3];
  const float* bk     = (const float*)d_in[4];
  const float* Wv     = (const float*)d_in[5];
  const float* bv     = (const float*)d_in[6];
  const float* Wsq    = (const float*)d_in[7];
  const float* bsq    = (const float*)d_in[8];
  const float* Wsk    = (const float*)d_in[9];
  const float* bsk    = (const float*)d_in[10];
  const float* Wsv    = (const float*)d_in[11];
  const float* bsv    = (const float*)d_in[12];
  const float* gammac = (const float*)d_in[13];
  const float* gammas = (const float*)d_in[14];

  char* ws = (char*)d_ws;
  // workspace layout (bytes), all 256B aligned
  __bf16* Wcat1 = (__bf16*)(ws + 0);                          // 1536*512*2   = 1572864
  __bf16* Wcat2 = (__bf16*)(ws + 1572864);                    // 640*512*2    = 655360
  float*  bias1 = (float*)(ws + 2228224);                     // 1536*4
  float*  bias2 = (float*)(ws + 2234368);                     // 640*4
  __bf16* xbt   = (__bf16*)(ws + 2236928);                    // B*N*C*2      = 16777216
  __bf16* Y1    = (__bf16*)(ws + 19014144);                   // B*1536*N*2   = 50331648 (q,k,sv)
  __bf16* Y2    = (__bf16*)(ws + 69345792);                   // B*N*640*2    = 20971520 (vT,sqT,skT)
  float*  E     = (float*)(ws + 90317312);                    // B*512*512*4  = 4194304
  __bf16* attn  = (__bf16*)(ws + 94511616);                   // B*512*512*2  = 2097152
  (void)ws_size; (void)n_in; (void)in_sizes; (void)out_size;

  // 1) weights -> bf16 concat + bias concat
  prep_weights<<<dim3((1536 * 512 + 640 * 512 + 255) / 256), 256, 0, stream>>>(
      Wq, Wk, Wsv, Wv, Wsq, Wsk, bq, bk, bsv, bv, bsq, bsk, Wcat1, Wcat2, bias1, bias2);

  // 2) xT bf16
  transpose_to_bf16<<<dim3(NN / 32, CC / 32, BB), dim3(32, 8), 0, stream>>>(x, xbt);

  // 3) Y1[b] = Wcat1 @ xT[b]^T + bias1   -> q,k,sv in [C][N]   (tiles 48x64 -> 384 blocks)
  gemm_nt<0><<<dim3(384, BB), 256, 0, stream>>>(
      Wcat1, xbt, Y1, 1536, NN, CC, CC, CC, NN,
      0LL, (long long)NN * CC, (long long)1536 * NN, bias1, 1, nullptr, nullptr);

  // 4) Y2[b] = xT[b] @ Wcat2^T + bias2   -> vT,sqT,skT in [N][·]  (tiles 128x10 -> 160 blocks)
  gemm_nt<0><<<dim3(160, BB), 256, 0, stream>>>(
      xbt, Wcat2, Y2, NN, 640, CC, CC, CC, 640,
      (long long)NN * CC, 0LL, (long long)NN * 640, bias2, 0, nullptr, nullptr);

  // 5) E[b] = q @ k^T  (f32)                                    (tiles 16x8 -> 16 blocks)
  gemm_nt<1><<<dim3(16, BB), 256, 0, stream>>>(
      Y1, Y1 + (size_t)512 * NN, E, 512, 512, NN, NN, NN, 512,
      (long long)1536 * NN, (long long)1536 * NN, (long long)512 * 512, nullptr, 0, nullptr, nullptr);

  // 6) attn = softmax_rows(E)  (bf16)
  softmax_rows<<<dim3(BB * 512 / 8), 256, 0, stream>>>(E, attn);

  // 7) d_out = gamma_c * (attn @ vT^T) + 2*x                     (tiles 16x64 -> 128 blocks)
  gemm_nt<2><<<dim3(128, BB), 256, 0, stream>>>(
      attn, Y2, d_out, 512, NN, 512, 512, 640, NN,
      (long long)512 * 512, (long long)NN * 640, (long long)512 * NN, nullptr, 0, x, gammac);

  // 8) d_out += gamma_s * spatial attention (two-pass flash over 4096 keys)
  flash_spatial<<<dim3(NN / 16, BB), 256, 0, stream>>>(Y2, Y1, gammas, (float*)d_out);
}
